// BiMultiHeadAttention_57140244906184
// MI455X (gfx1250) — matrix operand order repaired and checked by reading, not compiled, not run on hardware
//
#include <hip/hip_runtime.h>

// ---------------------------------------------------------------------------
// BiMultiHeadAttention (GLIP-style) for MI455X / gfx1250, wave32 + WMMA bf16.
//
// Shapes: B=4, H=4, Dh=256, E=1024, N_img=21760 (128^2+64^2+32^2+16^2),
//         N_txt=64, L=768.
// Heavy GEMMs: v_wmma_f32_16x16x32_bf16 with a 3-phase software pipeline per
// K-step (issue global loads -> WMMA current tile -> convert+commit to LDS),
// double-buffered LDS, global_prefetch_b8 lookahead.
// ---------------------------------------------------------------------------

typedef unsigned short bfr; // raw bf16 storage
typedef __attribute__((ext_vector_type(16))) __bf16 v16bf;
typedef __attribute__((ext_vector_type(8))) float v8f;
typedef __attribute__((ext_vector_type(4))) float f32x4;
typedef __attribute__((ext_vector_type(8))) unsigned short u16x8;
typedef __attribute__((ext_vector_type(16))) unsigned short u16x16;

__device__ __forceinline__ unsigned short f2bf(float f) {
  unsigned int u = __float_as_uint(f);
  u += 0x7FFFu + ((u >> 16) & 1u); // round-to-nearest-even
  return (unsigned short)(u >> 16);
}

template <typename T> __device__ __forceinline__ void store_val(T* p, float v);
template <> __device__ __forceinline__ void store_val<float>(float* p, float v) { *p = v; }
template <> __device__ __forceinline__ void store_val<unsigned short>(unsigned short* p, float v) { *p = f2bf(v); }

// ---------------------------------------------------------------------------
// Generic WMMA GEMM:  C[M,N] = alpha * A[M,K] x op(B) + bias
//   TRANSB=true : B is [N,K] row-major (x @ W^T form)
//   TRANSB=false: B is [K,N] row-major
// Workgroup: 256 thr = 8 waves. Tile 128(M) x 64(N) x 32(K), double-buffered.
// Wave tile 32x32 = 2x2 of v_wmma_f32_16x16x32_bf16.
// Output MODE: 0 = row-major (batch,sC,ldc), 1 = head-split bf16
//              ([B*rows, E] -> [(b*4+h)*rows + i][d]),
//              2 = head-merge ([bh][m][d] -> [b*rows+m][h*256+d]).
// ---------------------------------------------------------------------------
template <typename TA, typename TB, bool TRANSB, typename TC, int MODE>
__global__ __launch_bounds__(256) void gemm_wmma(
    const TA* __restrict__ A, const TB* __restrict__ B, TC* __restrict__ C,
    const float* __restrict__ bias, float alpha,
    int M, int N, int K, int lda, int ldb, int ldc,
    long long sA, long long sB, long long sC, int rows) {
  __shared__ alignas(16) unsigned short lA[2][128][40]; // 32 K + 8 pad (80B rows)
  __shared__ alignas(16) unsigned short lB[2][64][40];

  const int batch = blockIdx.z;
  A += (long long)batch * sA;
  B += (long long)batch * sB;

  const int tileM = blockIdx.y * 128;
  const int tileN = blockIdx.x * 64;
  const int tid = threadIdx.x;
  const int lane = tid & 31;
  const int wave = tid >> 5;
  const int wm = (wave & 3) * 32; // 4 waves along M
  const int wn = (wave >> 2) * 32; // 2 waves along N

  v8f acc[2][2] = {};

  const int aK0 = (lane < 16) ? 0 : 8;  // A frag: K groups {0-7,16-23} / {8-15,24-31}
  const int bK0 = (lane < 16) ? 0 : 8;  // B frag: K groups {0-7,16-23} / {8-15,24-31}
  const int fr = lane & 15;             // A: row-in-tile   B: col-in-tile

  // ---- per-thread staging geometry (fixed across K-steps) ----
  const int ar = tid >> 1, ac = (tid & 1) * 16;         // A: half-row per thread
  const bool aValid = (tileM + ar) < M;
  const long long aOff = (long long)(tileM + ar) * lda + ac;
  const int bn = tid >> 1, bc = (tid & 1) * 16;         // B NT: half-row per thread
  const int bkc = tid >> 3, bn0 = (tid & 7) * 8;        // B NN: (k, 8 n) per thread

  // staging registers (only the if-constexpr-live ones materialize)
  f32x4 raf[4]; u16x8 rau[2];
  f32x4 rbf[4]; u16x8 rbu[2];
  f32x4 rbnf[2]; u16x8 rbnu;

  auto loadA = [&](int kk) { // issue global loads (no conversion, no LDS)
    if (aValid) {
      const TA* src = A + aOff + kk;
      __builtin_prefetch(src + 64, 0, 1); // 2 K-steps ahead -> global_prefetch_b8
      if constexpr (sizeof(TA) == 4) {
        const f32x4* s4 = (const f32x4*)src;
        raf[0] = s4[0]; raf[1] = s4[1]; raf[2] = s4[2]; raf[3] = s4[3];
      } else {
        const u16x8* s8 = (const u16x8*)src;
        rau[0] = s8[0]; rau[1] = s8[1];
      }
    }
  };
  auto commitA = [&](int buf) { // convert + LDS store (wait lands here)
    u16x8 p0 = 0, p1 = 0;
    if (aValid) {
      if constexpr (sizeof(TA) == 4) {
#pragma unroll
        for (int j = 0; j < 4; ++j) {
          p0[j] = f2bf(raf[0][j]); p0[4 + j] = f2bf(raf[1][j]);
          p1[j] = f2bf(raf[2][j]); p1[4 + j] = f2bf(raf[3][j]);
        }
      } else {
        p0 = rau[0]; p1 = rau[1];
      }
    }
    *(u16x8*)&lA[buf][ar][ac] = p0;
    *(u16x8*)&lA[buf][ar][ac + 8] = p1;
  };

  auto loadB = [&](int kk) {
    if constexpr (TRANSB) {
      if (tid < 128) {
        const TB* src = B + (long long)(tileN + bn) * ldb + kk + bc;
        __builtin_prefetch(src + 64, 0, 1);
        if constexpr (sizeof(TB) == 4) {
          const f32x4* s4 = (const f32x4*)src;
          rbf[0] = s4[0]; rbf[1] = s4[1]; rbf[2] = s4[2]; rbf[3] = s4[3];
        } else {
          const u16x8* s8 = (const u16x8*)src;
          rbu[0] = s8[0]; rbu[1] = s8[1];
        }
      }
    } else {
      const TB* src = B + (long long)(kk + bkc) * ldb + tileN + bn0;
      if constexpr (sizeof(TB) == 4) {
        const f32x4* s4 = (const f32x4*)src;
        rbnf[0] = s4[0]; rbnf[1] = s4[1];
      } else {
        rbnu = *(const u16x8*)src;
      }
    }
  };
  auto commitB = [&](int buf) {
    if constexpr (TRANSB) {
      if (tid < 128) {
        u16x8 p0, p1;
        if constexpr (sizeof(TB) == 4) {
#pragma unroll
          for (int j = 0; j < 4; ++j) {
            p0[j] = f2bf(rbf[0][j]); p0[4 + j] = f2bf(rbf[1][j]);
            p1[j] = f2bf(rbf[2][j]); p1[4 + j] = f2bf(rbf[3][j]);
          }
        } else {
          p0 = rbu[0]; p1 = rbu[1];
        }
        *(u16x8*)&lB[buf][bn][bc] = p0;
        *(u16x8*)&lB[buf][bn][bc + 8] = p1;
      }
    } else {
      u16x8 p;
      if constexpr (sizeof(TB) == 4) {
#pragma unroll
        for (int j = 0; j < 4; ++j) { p[j] = f2bf(rbnf[0][j]); p[4 + j] = f2bf(rbnf[1][j]); }
      } else {
        p = rbnu;
      }
#pragma unroll
      for (int j = 0; j < 8; ++j) lB[buf][bn0 + j][bkc] = p[j];
    }
  };

  auto compute = [&](int buf) {
    v16bf af[2], bfv[2];
#pragma unroll
    for (int s = 0; s < 2; ++s) {
      u16x8 lo = *(const u16x8*)&lA[buf][wm + s * 16 + fr][aK0];
      u16x8 hi = *(const u16x8*)&lA[buf][wm + s * 16 + fr][aK0 + 16];
      u16x16 av;
#pragma unroll
      for (int j = 0; j < 8; ++j) { av[j] = lo[j]; av[8 + j] = hi[j]; }
      af[s] = __builtin_bit_cast(v16bf, av);
      u16x8 bl = *(const u16x8*)&lB[buf][wn + s * 16 + fr][bK0];
      u16x8 bh = *(const u16x8*)&lB[buf][wn + s * 16 + fr][bK0 + 16];
      u16x16 bv;
#pragma unroll
      for (int j = 0; j < 8; ++j) { bv[j] = bl[j]; bv[8 + j] = bh[j]; }
      bfv[s] = __builtin_bit_cast(v16bf, bv);
    }
#pragma unroll
    for (int sm = 0; sm < 2; ++sm)
#pragma unroll
      for (int sn = 0; sn < 2; ++sn)
        acc[sm][sn] = __builtin_amdgcn_wmma_f32_16x16x32_bf16(
            false, af[sm], false, bfv[sn], (short)0, acc[sm][sn], false, false);
  };

  // ---- software-pipelined main loop ----
  const int nk = K >> 5;
  loadA(0); loadB(0);
  commitA(0); commitB(0);
  __syncthreads();
  for (int i = 0; i < nk; ++i) {
    const int cur = i & 1;
    const bool more = (i + 1 < nk);
    if (more) { loadA((i + 1) << 5); loadB((i + 1) << 5); } // issue (no wait)
    compute(cur);                                            // wmma on tile i
    if (more) { commitA(cur ^ 1); commitB(cur ^ 1); }        // wait+convert+ds
    __syncthreads();
  }

  // ---- epilogue (C layout: VGPR r -> M = r (+8 for lanes 16-31), N = lane&15)
#pragma unroll
  for (int sm = 0; sm < 2; ++sm)
#pragma unroll
    for (int sn = 0; sn < 2; ++sn)
#pragma unroll
      for (int r = 0; r < 8; ++r) {
        int m = tileM + wm + sm * 16 + r + ((lane < 16) ? 0 : 8);
        int n = tileN + wn + sn * 16 + (lane & 15);
        if (m < M) {
          float v = acc[sm][sn][r] * alpha + (bias ? bias[n] : 0.0f);
          long long dst;
          if constexpr (MODE == 0) {
            dst = (long long)batch * sC + (long long)m * ldc + n;
          } else if constexpr (MODE == 1) { // head-split
            int b = m / rows, i2 = m - b * rows;
            int h = n >> 8, d = n & 255;
            dst = ((long long)((b * 4 + h) * rows + i2)) * 256 + d;
          } else { // head-merge
            int b = batch >> 2, h = batch & 3;
            dst = ((long long)(b * rows + m)) * 1024 + h * 256 + n;
          }
          store_val<TC>(C + dst, v);
        }
      }
}

// ---------------------------------------------------------------------------
__global__ void f32_to_bf16(const float* __restrict__ in, bfr* __restrict__ out, int n) {
  int i = blockIdx.x * blockDim.x + threadIdx.x;
  if (i < n) out[i] = f2bf(in[i]);
}

// 64->64ch 3x3 stride-2 pad-1 conv on the similarity maps (levels 1..3 output).
// aw layout: [bh][pix][t] (t=64 contiguous). One block (64 thr) per out pixel.
__global__ __launch_bounds__(64) void conv_down(
    const float* __restrict__ aw, const float* __restrict__ w,
    const float* __restrict__ b, float* __restrict__ convout) {
  __shared__ float patch[9][64];
  int bh = blockIdx.y;
  int p = blockIdx.x;
  int H, inOff, outOff, lop;
  if (p < 4096) { H = 128; inOff = 0; outOff = 0; lop = p; }
  else if (p < 5120) { H = 64; inOff = 16384; outOff = 4096; lop = p - 4096; }
  else { H = 32; inOff = 20480; outOff = 5120; lop = p - 5120; }
  int OW = H >> 1;
  int oy = lop / OW, ox = lop % OW;
  int t = threadIdx.x;
  const float* in = aw + ((long long)bh * 21760 + inOff) * 64;
#pragma unroll
  for (int kidx = 0; kidx < 9; ++kidx) {
    int ky = kidx / 3, kx = kidx % 3;
    int iy = 2 * oy - 1 + ky, ix = 2 * ox - 1 + kx;
    float v = 0.0f;
    if (iy >= 0 && iy < H && ix >= 0 && ix < H)
      v = in[((long long)iy * H + ix) * 64 + t];
    patch[kidx][t] = v;
  }
  __syncthreads();
  float sum = b[t];
  const float* wt = w + (long long)t * 576; // [O=64][I=64][3][3]
  for (int ci = 0; ci < 64; ++ci) {
#pragma unroll
    for (int kidx = 0; kidx < 9; ++kidx)
      sum = fmaf(patch[kidx][ci], wt[ci * 9 + kidx], sum);
  }
  convout[((long long)bh * 5376 + outOff + lop) * 64 + t] = sum;
}

// Pyramid update: aw2 = cur + 0.5*alpha*((bilinear_up(level+1)-cur) + (conv(level-1)-cur))
__global__ void sim_apply(const float* __restrict__ aw, const float* __restrict__ convout,
                          const float* __restrict__ alpha_p, float* __restrict__ aw2) {
  long long idx = (long long)blockIdx.x * blockDim.x + threadIdx.x;
  if (idx >= 16LL * 21760 * 64) return;
  int t = (int)(idx & 63);
  long long pg = idx >> 6;
  int bh = (int)(pg / 21760);
  int pix = (int)(pg % 21760);
  float alpha = alpha_p[0];
  int lvl, off, H;
  if (pix < 16384) { lvl = 0; off = 0; H = 128; }
  else if (pix < 20480) { lvl = 1; off = 16384; H = 64; }
  else if (pix < 21504) { lvl = 2; off = 20480; H = 32; }
  else { lvl = 3; off = 21504; H = 16; }
  int lp = pix - off;
  int y = lp / H, x = lp % H;
  const float* base = aw + (long long)bh * 21760 * 64;
  float cur = base[(long long)pix * 64 + t];
  float accv = 0.0f;
  if (lvl < 3) { // top-down: bilinear x2 upsample of coarser level, half-pixel centers
    int off2 = (lvl == 0) ? 16384 : (lvl == 1) ? 20480 : 21504;
    int H2 = H >> 1;
    float sy = y * 0.5f - 0.25f, sx = x * 0.5f - 0.25f;
    int y0 = (int)floorf(sy), x0 = (int)floorf(sx);
    float fy = sy - (float)y0, fx = sx - (float)x0;
    int y0c = min(max(y0, 0), H2 - 1), y1c = min(max(y0 + 1, 0), H2 - 1);
    int x0c = min(max(x0, 0), H2 - 1), x1c = min(max(x0 + 1, 0), H2 - 1);
    const float* l2 = base + (long long)off2 * 64 + t;
    float s00 = l2[((long long)y0c * H2 + x0c) * 64];
    float s01 = l2[((long long)y0c * H2 + x1c) * 64];
    float s10 = l2[((long long)y1c * H2 + x0c) * 64];
    float s11 = l2[((long long)y1c * H2 + x1c) * 64];
    float up = (1.0f - fy) * ((1.0f - fx) * s00 + fx * s01) +
               fy * ((1.0f - fx) * s10 + fx * s11);
    accv += alpha * (up - cur);
  }
  if (lvl > 0) { // bottom-up: stride-2 conv of finer level (resize is identity)
    int coff = (lvl == 1) ? 0 : (lvl == 2) ? 4096 : 5120;
    float cv = convout[((long long)bh * 5376 + coff + lp) * 64 + t];
    accv += alpha * (cv - cur);
  }
  aw2[((long long)bh * 21760 + pix) * 64 + t] = cur + 0.5f * accv;
}

// softmax over text dim (64): one wave per row, 2 elements/lane.
// (global-max subtract + clamp in reference is a softmax-invariant shift)
__global__ __launch_bounds__(256) void softmax_v(const float* __restrict__ aw2,
                                                 bfr* __restrict__ p_v, long long nrows) {
  long long row = (((long long)blockIdx.x * blockDim.x) + threadIdx.x) >> 5;
  int lane = threadIdx.x & 31;
  if (row >= nrows) return;
  const float* r = aw2 + row * 64;
  float a = r[lane], b = r[lane + 32];
  float m = fmaxf(a, b);
#pragma unroll
  for (int s = 16; s > 0; s >>= 1) m = fmaxf(m, __shfl_xor(m, s, 32));
  float ea = __expf(a - m), eb = __expf(b - m);
  float sum = ea + eb;
#pragma unroll
  for (int s = 16; s > 0; s >>= 1) sum += __shfl_xor(sum, s, 32);
  float inv = 1.0f / sum;
  p_v[row * 64 + lane] = f2bf(ea * inv);
  p_v[row * 64 + lane + 32] = f2bf(eb * inv);
}

// softmax over image dim (21760): one 256-thr block per (bh, t) row.
__global__ __launch_bounds__(256) void softmax_l(const float* __restrict__ aw2,
                                                 bfr* __restrict__ p_l) {
  int bh = blockIdx.x >> 6, t = blockIdx.x & 63;
  const float* base = aw2 + (long long)bh * 21760 * 64 + t;
  __shared__ float red[256];
  int tid = threadIdx.x;
  float m = -3.0e38f;
  for (int i = tid; i < 21760; i += 256) m = fmaxf(m, base[(long long)i * 64]);
  red[tid] = m;
  __syncthreads();
  for (int s = 128; s > 0; s >>= 1) {
    if (tid < s) red[tid] = fmaxf(red[tid], red[tid + s]);
    __syncthreads();
  }
  m = red[0];
  __syncthreads();
  float sum = 0.0f;
  for (int i = tid; i < 21760; i += 256) sum += __expf(base[(long long)i * 64] - m);
  red[tid] = sum;
  __syncthreads();
  for (int s = 128; s > 0; s >>= 1) {
    if (tid < s) red[tid] += red[tid + s];
    __syncthreads();
  }
  float inv = 1.0f / red[0];
  bfr* out = p_l + ((long long)bh * 64 + t) * 21760;
  for (int i = tid; i < 21760; i += 256)
    out[i] = f2bf(__expf(base[(long long)i * 64] - m) * inv);
}

// ---------------------------------------------------------------------------
extern "C" void kernel_launch(void* const* d_in, const int* in_sizes, int n_in,
                              void* d_out, int out_size, void* d_ws, size_t ws_size,
                              hipStream_t stream) {
  (void)in_sizes; (void)n_in; (void)out_size; (void)ws_size;
  const float* v = (const float*)d_in[0];
  const float* l = (const float*)d_in[1];
  const float* sim_parm = (const float*)d_in[2];
  const float* vp_w = (const float*)d_in[3];
  const float* vp_b = (const float*)d_in[4];
  const float* lp_w = (const float*)d_in[5];
  const float* lp_b = (const float*)d_in[6];
  const float* vvp_w = (const float*)d_in[7];
  const float* vvp_b = (const float*)d_in[8];
  const float* vlp_w = (const float*)d_in[9];
  const float* vlp_b = (const float*)d_in[10];
  const float* ovp_w = (const float*)d_in[11];
  const float* ovp_b = (const float*)d_in[12];
  const float* olp_w = (const float*)d_in[13];
  const float* olp_b = (const float*)d_in[14];
  const float* down_w = (const float*)d_in[15];
  const float* down_b = (const float*)d_in[16];
  float* outp = (float*)d_out;

  // ---- workspace bump allocator (peak ~565 MB) ----
  char* ws = (char*)d_ws;
  size_t off = 0;
  auto alloc = [&](size_t bytes) {
    char* p = ws + off;
    off += (bytes + 255) & ~(size_t)255;
    return p;
  };
  bfr* wvp  = (bfr*)alloc(1024 * 256 * 2);
  bfr* wlp  = (bfr*)alloc(1024 * 768 * 2);
  bfr* wvvp = (bfr*)alloc(1024 * 256 * 2);
  bfr* wvlp = (bfr*)alloc(1024 * 768 * 2);
  bfr* wovp = (bfr*)alloc(256 * 1024 * 2);
  bfr* wolp = (bfr*)alloc(768 * 1024 * 2);
  bfr* qb   = (bfr*)alloc(16LL * 21760 * 256 * 2);  // later reused as out_v_bf
  bfr* vvb  = (bfr*)alloc(16LL * 21760 * 256 * 2);
  bfr* kb   = (bfr*)alloc(16LL * 64 * 256 * 2);
  bfr* vlb  = (bfr*)alloc(16LL * 64 * 256 * 2);
  float* aw = (float*)alloc(16LL * 21760 * 64 * 4); // later reused for p_v/p_l
  float* aw2 = (float*)alloc(16LL * 21760 * 64 * 4);
  float* convout = (float*)alloc(16LL * 5376 * 64 * 4);
  bfr* out_l_bf = (bfr*)alloc(4LL * 64 * 1024 * 2);
  bfr* p_v = (bfr*)aw;                       // alias: aw dead after sim_apply
  bfr* p_l = p_v + 16LL * 21760 * 64;
  bfr* out_v_bf = qb;                        // alias: q dead after logits

  // ---- 1) weights -> bf16 ----
  auto cvt = [&](const float* s, bfr* d, int n) {
    f32_to_bf16<<<dim3((n + 255) / 256), dim3(256), 0, stream>>>(s, d, n);
  };
  cvt(vp_w, wvp, 262144);
  cvt(lp_w, wlp, 786432);
  cvt(vvp_w, wvvp, 262144);
  cvt(vlp_w, wvlp, 786432);
  cvt(ovp_w, wovp, 262144);
  cvt(olp_w, wolp, 786432);

  // ---- 2) input projections (head-split epilogue) ----
  // q = (v @ vp_w^T) * SCALE + b ; SCALE = 256^-0.5 = 0.0625
  gemm_wmma<float, bfr, true, bfr, 1><<<dim3(16, 680, 1), 256, 0, stream>>>(
      v, wvp, qb, vp_b, 0.0625f, 87040, 1024, 256, 256, 256, 0, 0, 0, 0, 21760);
  gemm_wmma<float, bfr, true, bfr, 1><<<dim3(16, 680, 1), 256, 0, stream>>>(
      v, wvvp, vvb, vvp_b, 1.0f, 87040, 1024, 256, 256, 256, 0, 0, 0, 0, 21760);
  gemm_wmma<float, bfr, true, bfr, 1><<<dim3(16, 2, 1), 256, 0, stream>>>(
      l, wlp, kb, lp_b, 1.0f, 256, 1024, 768, 768, 768, 0, 0, 0, 0, 64);
  gemm_wmma<float, bfr, true, bfr, 1><<<dim3(16, 2, 1), 256, 0, stream>>>(
      l, wvlp, vlb, vlp_b, 1.0f, 256, 1024, 768, 768, 768, 0, 0, 0, 0, 64);

  // ---- 3) logits aw[bh] = q[bh] @ k[bh]^T  (batched NT) ----
  gemm_wmma<bfr, bfr, true, float, 0><<<dim3(1, 170, 16), 256, 0, stream>>>(
      qb, kb, aw, nullptr, 1.0f, 21760, 64, 256, 256, 256, 64,
      21760LL * 256, 64LL * 256, 21760LL * 64, 0);

  // ---- 4) similarity_update pyramid ----
  conv_down<<<dim3(5376, 16), dim3(64), 0, stream>>>(aw, down_w, down_b, convout);
  sim_apply<<<dim3(87040), dim3(256), 0, stream>>>(aw, convout, sim_parm, aw2);

  // ---- 5) softmaxes ----
  softmax_v<<<dim3(43520), dim3(256), 0, stream>>>(aw2, p_v, 348160LL);
  softmax_l<<<dim3(1024), dim3(256), 0, stream>>>(aw2, p_l);

  // ---- 6) attention outputs (batched NN, head-merge epilogue) ----
  // out_v[bh] = p_v[bh] @ val_l[bh]   -> [4, 21760, 1024] bf16
  gemm_wmma<bfr, bfr, false, bfr, 2><<<dim3(4, 170, 16), 256, 0, stream>>>(
      p_v, vlb, out_v_bf, nullptr, 1.0f, 21760, 256, 64, 64, 256, 0,
      21760LL * 64, 64LL * 256, 0, 21760);
  // out_l[bh] = p_l[bh] @ val_v[bh]   -> [4, 64, 1024] bf16
  gemm_wmma<bfr, bfr, false, bfr, 2><<<dim3(4, 1, 16), 256, 0, stream>>>(
      p_l, vvb, out_l_bf, nullptr, 1.0f, 64, 256, 21760, 21760, 256, 0,
      64LL * 21760, 21760LL * 256, 0, 64);

  // ---- 7) output projections (f32 out to d_out) ----
  gemm_wmma<bfr, bfr, true, float, 0><<<dim3(4, 680, 1), 256, 0, stream>>>(
      out_v_bf, wovp, outp, ovp_b, 1.0f, 87040, 256, 1024, 1024, 1024, 256,
      0, 0, 0, 0);
  gemm_wmma<bfr, bfr, true, float, 0><<<dim3(12, 2, 1), 256, 0, stream>>>(
      out_l_bf, wolp, outp + 22282240LL, olp_b, 1.0f, 256, 768, 1024, 1024, 1024, 768,
      0, 0, 0, 0);
}
